// Decoder_40132174414202
// MI455X (gfx1250) — matrix operand order repaired and checked by reading, not compiled, not run on hardware
//
#include <hip/hip_runtime.h>
#include <hip/hip_bf16.h>

// ---------------------------------------------------------------------------
// Tacotron2 decoder for MI455X (gfx1250, wave32, WMMA bf16 16x16x32)
// N-blocked WMMA GEMM: 4 N-tiles per wave -> A-fragment loaded once per
// k-step, 4 independent v_wmma back-to-back (no D->A/B RAW hazards).
// ---------------------------------------------------------------------------

typedef __attribute__((ext_vector_type(16))) __bf16 bfx16;
typedef __attribute__((ext_vector_type(8)))  float  fx8;

#define DB    32
#define DT    512
#define DE    512
#define DTDEC 400
#define DNMEL 80
#define DRNN  1024
#define DPRE  256
#define DATT  128
#define DG    4096
#define KATT  1792   // PRE + E + ATT_RNN
#define KDEC  2560   // ATT_RNN + E + DEC_RNN
#define KP1   96     // NMEL (80) zero-padded to multiple of 32

__device__ __forceinline__ __bf16 f2bf(float x) {
  union { float f; unsigned u; } v; v.f = x;
  unsigned r = v.u + 0x7FFFu + ((v.u >> 16) & 1u);
  union { unsigned short s; __bf16 b; } o; o.s = (unsigned short)(r >> 16);
  return o.b;
}
__device__ __forceinline__ float bf2f(__bf16 x) {
  union { __bf16 b; unsigned short s; } v; v.b = x;
  union { unsigned u; float f; } o; o.u = ((unsigned)v.s) << 16;
  return o.f;
}
__device__ __forceinline__ float sigm(float x) { return 1.f / (1.f + __expf(-x)); }

// -------------------- WMMA bf16 GEMM: C[M,N] = A[M,K] * W[N,K]^T -----------
// A fragment 16x32: lanes 0-15 -> M=lane, K {0..7,16..23}; lanes 16-31 -> K {8..15,24..31}
// B fragment 32x16: lane n = lane&15, same K order as A: K {0..7,16..23} / {8..15,24..31}.
// NB = N-tiles per wave (register blocking; A loaded once, NB wmma per k-step).
template <int NB>
__global__ void dec_gemm(const __bf16* __restrict__ A, int lda,
                         const __bf16* __restrict__ W, int ldw,
                         float* __restrict__ Cf, __bf16* __restrict__ Cb,
                         long rs, long cs,            // output row/col strides
                         const float* __restrict__ bias,
                         int numM, int numN, int K, int relu) {
  const int wid  = (blockIdx.x * blockDim.x + threadIdx.x) >> 5;
  const int nw   = (gridDim.x * blockDim.x) >> 5;
  const int lane = threadIdx.x & 31;
  const int am   = lane & 15;
  const int akb  = (lane >> 4) * 8;
  const int bn   = lane & 15;
  const int bkb  = (lane >> 4) * 8;
  const int ngrp  = numN / (NB * 16);
  const int tiles = (numM / 16) * ngrp;
  for (int tl = wid; tl < tiles; tl += nw) {
    const int mt = tl / ngrp, ng = tl % ngrp;
    fx8 acc[NB];
#pragma unroll
    for (int j = 0; j < NB; ++j) acc[j] = fx8{};
    const __bf16* arow = A + (size_t)(mt * 16 + am) * lda + akb;
    const __bf16* brow = W + (size_t)(ng * NB * 16 + bn) * ldw + bkb;
#pragma unroll 2
    for (int k = 0; k < K; k += 32) {
      union { uint4 q[2]; bfx16 v; } af;
      af.q[0] = *(const uint4*)(arow + k);        // K = k+akb    .. +7
      af.q[1] = *(const uint4*)(arow + k + 16);   // K = k+16+akb .. +7
#pragma unroll
      for (int j = 0; j < NB; ++j) {
        union { uint4 q[2]; bfx16 v; } bf;
        const __bf16* bp = brow + (size_t)j * 16 * ldw + k;
        bf.q[0] = *(const uint4*)(bp);            // K = k+bkb    .. +7
        bf.q[1] = *(const uint4*)(bp + 16);       // K = k+16+bkb .. +7
        acc[j] = __builtin_amdgcn_wmma_f32_16x16x32_bf16(false, af.v, false, bf.v,
                                                         (short)0, acc[j], false, false);
      }
    }
    const int m0 = mt * 16 + (lane >> 4) * 8;
#pragma unroll
    for (int j = 0; j < NB; ++j) {
      const int n = (ng * NB + j) * 16 + (lane & 15);
      const float bv = bias ? bias[n] : 0.f;
#pragma unroll
      for (int v = 0; v < 8; ++v) {
        float val = acc[j][v] + bv;
        if (relu) val = fmaxf(val, 0.f);
        const size_t idx = (size_t)(m0 + v) * rs + (size_t)n * cs;
        if (Cf) Cf[idx] = val;
        if (Cb) Cb[idx] = f2bf(val);
      }
    }
  }
}

// -------------------- elementwise / prep kernels ---------------------------
__global__ void dec_zero(unsigned* p, long n) {
  for (long i = blockIdx.x * (long)blockDim.x + threadIdx.x; i < n;
       i += (long)gridDim.x * blockDim.x) p[i] = 0u;
}
__global__ void dec_cvt(const float* __restrict__ s, __bf16* __restrict__ d, long n) {
  for (long i = blockIdx.x * (long)blockDim.x + threadIdx.x; i < n;
       i += (long)gridDim.x * blockDim.x) d[i] = f2bf(s[i]);
}
__global__ void dec_padcvt(const float* __restrict__ s, int K,
                           __bf16* __restrict__ d, int Kp, long rows) {
  const long n = rows * Kp;
  for (long i = blockIdx.x * (long)blockDim.x + threadIdx.x; i < n;
       i += (long)gridDim.x * blockDim.x) {
    long r = i / Kp; int k = (int)(i % Kp);
    d[i] = f2bf(k < K ? s[r * K + k] : 0.f);
  }
}
__global__ void dec_concat(const float* __restrict__ A, int Ka,
                           const float* __restrict__ B2, int Kb,
                           __bf16* __restrict__ d, long total) {
  const int Kt = Ka + Kb;
  for (long i = blockIdx.x * (long)blockDim.x + threadIdx.x; i < total;
       i += (long)gridDim.x * blockDim.x) {
    long n = i / Kt; int k = (int)(i % Kt);
    float v = (k < Ka) ? A[n * Ka + k] : B2[n * Kb + (k - Ka)];
    d[i] = f2bf(v);
  }
}
__global__ void dec_badd(const float* a, const float* b, float* c, int n) {
  int i = blockIdx.x * blockDim.x + threadIdx.x;
  if (i < n) c[i] = a[i] + b[i];
}
// pin (prenet input, teacher forced & shifted), zero-padded K 80->96, bf16
__global__ void dec_build_pin(const float* __restrict__ din, __bf16* __restrict__ d) {
  const long n = (long)DTDEC * DB * KP1;
  for (long i = blockIdx.x * (long)blockDim.x + threadIdx.x; i < n;
       i += (long)gridDim.x * blockDim.x) {
    long r = i / KP1; int k = (int)(i % KP1);
    int t = (int)(r / DB), b = (int)(r % DB);
    float v = 0.f;
    if (k < DNMEL && t > 0) v = din[((size_t)b * DNMEL + k) * DTDEC + (t - 1)];
    d[i] = f2bf(v);
  }
}

// -------------------- per-step kernels -------------------------------------
// xh_att = [ pre_t | ctx | ah ]  (bf16, K = 1792)
__global__ void dec_pack_att(const __bf16* __restrict__ pre_t, const float* __restrict__ ctx,
                             const __bf16* __restrict__ ah_bf, __bf16* __restrict__ xh) {
  const long n = (long)DB * KATT;
  for (long i = blockIdx.x * (long)blockDim.x + threadIdx.x; i < n;
       i += (long)gridDim.x * blockDim.x) {
    int b = (int)(i / KATT), k = (int)(i % KATT);
    __bf16 v;
    if (k < DPRE)            v = pre_t[b * DPRE + k];
    else if (k < DPRE + DE)  v = f2bf(ctx[b * DE + (k - DPRE)]);
    else                     v = ah_bf[b * DRNN + (k - DPRE - DE)];
    xh[i] = v;
  }
}
// xh_dec = [ ah | ctx | dh ] (bf16, K=2560); also fills dhc[:,1024:1536] = ctx bf16
__global__ void dec_pack_dec(const __bf16* __restrict__ ah_bf, const float* __restrict__ ctx,
                             __bf16* __restrict__ dhc, __bf16* __restrict__ xh) {
  const long n = (long)DB * KDEC;
  for (long i = blockIdx.x * (long)blockDim.x + threadIdx.x; i < n;
       i += (long)gridDim.x * blockDim.x) {
    int b = (int)(i / KDEC), k = (int)(i % KDEC);
    __bf16 v;
    if (k < DRNN)            v = ah_bf[b * DRNN + k];
    else if (k < DRNN + DE) {
      v = f2bf(ctx[b * DE + (k - DRNN)]);
      dhc[b * (DRNN + DE) + k] = v;                 // ctx half of dhc
    } else                   v = dhc[b * (DRNN + DE) + (k - DRNN - DE)]; // prev dh
    xh[i] = v;
  }
}
// LSTM pointwise: gates [32,4096] f32 -> c (in/out f32), h (bf16, ldh)
__global__ void dec_lstm_act(const float* __restrict__ g, float* __restrict__ c,
                             __bf16* __restrict__ hb, int ldh) {
  const long n = (long)DB * DRNN;
  for (long i = blockIdx.x * (long)blockDim.x + threadIdx.x; i < n;
       i += (long)gridDim.x * blockDim.x) {
    int b = (int)(i / DRNN), j = (int)(i % DRNN);
    const float* gr = g + (size_t)b * DG;
    float gi = gr[j], gf = gr[DRNN + j], gg = gr[2 * DRNN + j], go = gr[3 * DRNN + j];
    float c2 = sigm(gf) * c[i] + sigm(gi) * tanhf(gg);
    c[i] = c2;
    hb[(size_t)b * ldh + j] = f2bf(sigm(go) * tanhf(c2));
  }
}

// Location-sensitive attention: one block per batch row, one thread per t.
__global__ void __launch_bounds__(DT)
dec_attention(float* __restrict__ aw, float* __restrict__ awcum,
              const float* __restrict__ pm, const float* __restrict__ q,
              const float* __restrict__ mem,
              const float* __restrict__ wconv, const float* __restrict__ wdense,
              const float* __restrict__ wscore, const float* __restrict__ bscore,
              const unsigned char* __restrict__ mask,
              float* __restrict__ ctx, float* __restrict__ align_out, int tstep) {
  __shared__ float s_aw[DT], s_awc[DT], s_w[DT], s_red[DT];
  __shared__ float s_q[DATT], s_ws[DATT];
  __shared__ float s_wc[31 * 62];       // [f][c][k]
  __shared__ float s_wd[DATT * 31];     // [a][f]
  const int b = blockIdx.x, t = threadIdx.x;

  s_aw[t]  = aw[b * DT + t];
  s_awc[t] = awcum[b * DT + t];
  for (int i = t; i < 31 * 62;   i += DT) s_wc[i] = wconv[i];
  for (int i = t; i < DATT * 31; i += DT) s_wd[i] = wdense[i];
  if (t < DATT) { s_q[t] = q[b * DATT + t]; s_ws[t] = wscore[t]; }
  __syncthreads();

  // 1D conv over [aw, awcum], 31 filters, kernel 31, pad 15
  float loc[31];
#pragma unroll
  for (int f = 0; f < 31; ++f) {
    float a = 0.f;
    for (int k = 0; k < 31; ++k) {
      int tt = t + k - 15;
      if (tt >= 0 && tt < DT)
        a += s_aw[tt] * s_wc[f * 62 + k] + s_awc[tt] * s_wc[f * 62 + 31 + k];
    }
    loc[f] = a;
  }
  // score = sum_a tanh(q_a + loc@Wd^T + pm) * Ws
  const float* pmrow = pm + ((size_t)b * DT + t) * DATT;
  float sc = 0.f;
  for (int a = 0; a < DATT; ++a) {
    float ld = 0.f;
#pragma unroll
    for (int f = 0; f < 31; ++f) ld += loc[f] * s_wd[a * 31 + f];
    sc += tanhf(s_q[a] + ld + pmrow[a]) * s_ws[a];
  }
  sc += bscore[0];
  if (mask[b * DT + t]) sc = -1e9f;

  // softmax over t (block-wide)
  s_red[t] = sc; __syncthreads();
  for (int s = DT / 2; s > 0; s >>= 1) {
    if (t < s) s_red[t] = fmaxf(s_red[t], s_red[t + s]);
    __syncthreads();
  }
  const float mx = s_red[0]; __syncthreads();
  const float e = __expf(sc - mx);
  s_red[t] = e; __syncthreads();
  for (int s = DT / 2; s > 0; s >>= 1) {
    if (t < s) s_red[t] += s_red[t + s];
    __syncthreads();
  }
  const float w = e / s_red[0];

  aw[b * DT + t] = w;
  awcum[b * DT + t] = s_awc[t] + w;
  align_out[((size_t)b * DTDEC + tstep) * DT + t] = w;
  s_w[t] = w; __syncthreads();

  // ctx[b,e] = sum_tau w[tau] * memory[b,tau,e]   (thread = e, coalesced)
  const float* memb = mem + (size_t)b * DT * DE + t;
  float acc = 0.f;
  for (int tau = 0; tau < DT; ++tau) acc += s_w[tau] * memb[(size_t)tau * DE];
  ctx[b * DE + t] = acc;
}

// gate dot product (mel already written by strided GEMM epilogue)
__global__ void dec_gate(const __bf16* __restrict__ dhc,
                         const float* __restrict__ wgate, const float* __restrict__ bgate,
                         float* __restrict__ gate_out, int tstep) {
  __shared__ float red[128];
  const int b = blockIdx.x, tid = threadIdx.x;
  float p = 0.f;
  for (int j = tid; j < DRNN + DE; j += 128)
    p += bf2f(dhc[(size_t)b * (DRNN + DE) + j]) * wgate[j];
  red[tid] = p; __syncthreads();
  for (int s = 64; s > 0; s >>= 1) {
    if (tid < s) red[tid] += red[tid + s];
    __syncthreads();
  }
  if (tid == 0) gate_out[b * DTDEC + tstep] = red[0] + bgate[0];
}

// ---------------------------------------------------------------------------
extern "C" void kernel_launch(void* const* d_in, const int* in_sizes, int n_in,
                              void* d_out, int out_size, void* d_ws, size_t ws_size,
                              hipStream_t stream) {
  (void)in_sizes; (void)n_in; (void)out_size; (void)ws_size;
  const float* memory   = (const float*)d_in[0];
  const float* din      = (const float*)d_in[1];
  const float* W_pre1   = (const float*)d_in[2];
  const float* W_pre2   = (const float*)d_in[3];
  const float* Wih_att  = (const float*)d_in[4];
  const float* Whh_att  = (const float*)d_in[5];
  const float* bih_att  = (const float*)d_in[6];
  const float* bhh_att  = (const float*)d_in[7];
  const float* Wih_dec  = (const float*)d_in[8];
  const float* Whh_dec  = (const float*)d_in[9];
  const float* bih_dec  = (const float*)d_in[10];
  const float* bhh_dec  = (const float*)d_in[11];
  const float* W_query  = (const float*)d_in[12];
  const float* W_memory = (const float*)d_in[13];
  const float* W_score  = (const float*)d_in[14];
  const float* b_score  = (const float*)d_in[15];
  const float* W_lconv  = (const float*)d_in[16];
  const float* W_ldense = (const float*)d_in[17];
  const float* W_proj   = (const float*)d_in[18];
  const float* b_proj   = (const float*)d_in[19];
  const float* W_gate   = (const float*)d_in[20];
  const float* b_gate   = (const float*)d_in[21];
  const unsigned char* mask = (const unsigned char*)d_in[22];

  float* mel_out   = (float*)d_out;
  float* gate_out  = mel_out + (size_t)DB * DNMEL * DTDEC;
  float* align_out = gate_out + (size_t)DB * DTDEC;

  // ---- workspace carve-out (256B aligned slabs) ----
  char* wsp = (char*)d_ws;
  auto take = [&](size_t bytes) {
    char* p = wsp; wsp += (bytes + 255) & ~(size_t)255; return p;
  };
  __bf16* Wcat_att = (__bf16*)take((size_t)DG * KATT * 2);
  __bf16* Wcat_dec = (__bf16*)take((size_t)DG * KDEC * 2);
  __bf16* Wq_bf    = (__bf16*)take((size_t)DATT * DRNN * 2);
  __bf16* Wproj_bf = (__bf16*)take((size_t)DNMEL * (DRNN + DE) * 2);
  __bf16* Wmem_bf  = (__bf16*)take((size_t)DATT * DE * 2);
  __bf16* Wpre1p   = (__bf16*)take((size_t)DPRE * KP1 * 2);
  __bf16* Wpre2_bf = (__bf16*)take((size_t)DPRE * DPRE * 2);
  __bf16* mem_bf   = (__bf16*)take((size_t)DB * DT * DE * 2);
  float*  pmbuf    = (float*) take((size_t)DB * DT * DATT * 4);
  __bf16* pinp     = (__bf16*)take((size_t)DTDEC * DB * KP1 * 2);
  __bf16* pre1     = (__bf16*)take((size_t)DTDEC * DB * DPRE * 2);
  __bf16* preb     = (__bf16*)take((size_t)DTDEC * DB * DPRE * 2);
  float*  bsum_att = (float*) take(DG * 4);
  float*  bsum_dec = (float*) take(DG * 4);
  float*  gates    = (float*) take((size_t)DB * DG * 4);
  __bf16* xh_att   = (__bf16*)take((size_t)DB * KATT * 2);
  __bf16* xh_dec   = (__bf16*)take((size_t)DB * KDEC * 2);
  float*  qbuf     = (float*) take((size_t)DB * DATT * 4);
  // recurrent state (zeroed every launch -> deterministic)
  char* stateBase = wsp;
  float*  ac     = (float*) take((size_t)DB * DRNN * 4);
  float*  dc     = (float*) take((size_t)DB * DRNN * 4);
  float*  ctx    = (float*) take((size_t)DB * DE * 4);
  float*  awv    = (float*) take((size_t)DB * DT * 4);
  float*  awcum  = (float*) take((size_t)DB * DT * 4);
  __bf16* ah_bf  = (__bf16*)take((size_t)DB * DRNN * 2);
  __bf16* dhc_bf = (__bf16*)take((size_t)DB * (DRNN + DE) * 2);
  const long stateWords = (long)((wsp - stateBase) / 4);

  auto gs = [](long n, int tpb) {
    long bl = (n + tpb - 1) / tpb; return (int)(bl > 4096 ? 4096 : bl);
  };
  auto gb = [](long tiles) {
    long bl = (tiles + 3) / 4; if (bl < 1) bl = 1; return (int)(bl > 4096 ? 4096 : bl);
  };
  // NB=4 blocked launch when numN divisible by 64, else 1 tile per wave
  auto launch_gemm = [&](const __bf16* A, int lda, const __bf16* W, int ldw,
                         float* Cf, __bf16* Cb, long rs, long cs,
                         const float* bias, int M, int N, int K, int relu) {
    if (N % 64 == 0) {
      dec_gemm<4><<<gb((long)(M / 16) * (N / 64)), 128, 0, stream>>>(
          A, lda, W, ldw, Cf, Cb, rs, cs, bias, M, N, K, relu);
    } else {
      dec_gemm<1><<<gb((long)(M / 16) * (N / 16)), 128, 0, stream>>>(
          A, lda, W, ldw, Cf, Cb, rs, cs, bias, M, N, K, relu);
    }
  };

  // ---- one-time prep ----
  dec_zero<<<gs(stateWords, 256), 256, 0, stream>>>((unsigned*)stateBase, stateWords);
  dec_cvt<<<gs((long)DATT * DRNN, 256), 256, 0, stream>>>(W_query, Wq_bf, (long)DATT * DRNN);
  dec_cvt<<<gs((long)DNMEL * (DRNN + DE), 256), 256, 0, stream>>>(W_proj, Wproj_bf, (long)DNMEL * (DRNN + DE));
  dec_cvt<<<gs((long)DATT * DE, 256), 256, 0, stream>>>(W_memory, Wmem_bf, (long)DATT * DE);
  dec_cvt<<<gs((long)DPRE * DPRE, 256), 256, 0, stream>>>(W_pre2, Wpre2_bf, (long)DPRE * DPRE);
  dec_cvt<<<gs((long)DB * DT * DE, 256), 256, 0, stream>>>(memory, mem_bf, (long)DB * DT * DE);
  dec_padcvt<<<gs((long)DPRE * KP1, 256), 256, 0, stream>>>(W_pre1, DNMEL, Wpre1p, KP1, DPRE);
  dec_concat<<<gs((long)DG * KATT, 256), 256, 0, stream>>>(Wih_att, DPRE + DE, Whh_att, DRNN, Wcat_att, (long)DG * KATT);
  dec_concat<<<gs((long)DG * KDEC, 256), 256, 0, stream>>>(Wih_dec, DRNN + DE, Whh_dec, DRNN, Wcat_dec, (long)DG * KDEC);
  dec_badd<<<(DG + 255) / 256, 256, 0, stream>>>(bih_att, bhh_att, bsum_att, DG);
  dec_badd<<<(DG + 255) / 256, 256, 0, stream>>>(bih_dec, bhh_dec, bsum_dec, DG);
  dec_build_pin<<<gs((long)DTDEC * DB * KP1, 256), 256, 0, stream>>>(din, pinp);

  // prenet over ALL 400 steps at once (two big WMMA GEMMs, relu, bf16 out)
  const int Mrows = DTDEC * DB;                 // 12800
  launch_gemm(pinp, KP1, Wpre1p, KP1, nullptr, pre1, DPRE, 1, nullptr,
              Mrows, DPRE, KP1, 1);
  launch_gemm(pre1, DPRE, Wpre2_bf, DPRE, nullptr, preb, DPRE, 1, nullptr,
              Mrows, DPRE, DPRE, 1);
  // processed_memory = memory @ W_memory^T  (M=16384, N=128, K=512)
  launch_gemm(mem_bf, DE, Wmem_bf, DE, pmbuf, nullptr, DATT, 1, nullptr,
              DB * DT, DATT, DE, 0);

  // ---- 400 sequential decoder steps ----
  for (int t = 0; t < DTDEC; ++t) {
    dec_pack_att<<<gs((long)DB * KATT, 256), 256, 0, stream>>>(
        preb + (size_t)t * DB * DPRE, ctx, ah_bf, xh_att);
    launch_gemm(xh_att, KATT, Wcat_att, KATT, gates, nullptr, DG, 1, bsum_att,
                DB, DG, KATT, 0);                                   // att-LSTM gates
    dec_lstm_act<<<gs((long)DB * DRNN, 256), 256, 0, stream>>>(gates, ac, ah_bf, DRNN);
    launch_gemm(ah_bf, DRNN, Wq_bf, DRNN, qbuf, nullptr, DATT, 1, nullptr,
                DB, DATT, DRNN, 0);                                 // query projection
    dec_attention<<<DB, DT, 0, stream>>>(awv, awcum, pmbuf, qbuf, memory,
                                         W_lconv, W_ldense, W_score, b_score,
                                         mask, ctx, align_out, t);
    dec_pack_dec<<<gs((long)DB * KDEC, 256), 256, 0, stream>>>(ah_bf, ctx, dhc_bf, xh_dec);
    launch_gemm(xh_dec, KDEC, Wcat_dec, KDEC, gates, nullptr, DG, 1, bsum_dec,
                DB, DG, KDEC, 0);                                   // dec-LSTM gates
    dec_lstm_act<<<gs((long)DB * DRNN, 256), 256, 0, stream>>>(gates, dc, dhc_bf, DRNN + DE);
    // mel projection written straight into [B,NMEL,TDEC] output layout
    launch_gemm(dhc_bf, DRNN + DE, Wproj_bf, DRNN + DE,
                mel_out + t, nullptr, (long)DNMEL * DTDEC, DTDEC, b_proj,
                DB, DNMEL, DRNN + DE, 0);
    dec_gate<<<DB, 128, 0, stream>>>(dhc_bf, W_gate, b_gate, gate_out, t);
  }
}
